// GCFAgg_75668733821313
// MI455X (gfx1250) — matrix operand order repaired and checked by reading, not compiled, not run on hardware
//
#include <hip/hip_runtime.h>
#include <stddef.h>

// ---------------------------------------------------------------------------
// GCFAgg-style fused kernel set for MI455X (gfx1250, wave32, WMMA bf16)
//   q1 = x@Wq1^T + b1 ; q2 = x@Wq2^T + b2 ; r = x@WR^T + bR   (bf16 in ws)
//   out = (q1 @ q2^T) @ r     fused, s never leaves the WGP
// CDNA5 paths used: v_wmma_f32_16x16x32_bf16, global_load_async_to_lds_b128
// (ASYNCcnt), s_wait_asynccnt, global_prefetch_b8.
// ---------------------------------------------------------------------------

typedef __attribute__((ext_vector_type(16))) __bf16 v16bf;
typedef __attribute__((ext_vector_type(8)))  float  v8f;

union BFrag {
    v16bf v;
    unsigned short h[16];
};

__device__ __forceinline__ unsigned short f2bf(float f) {
    union { float f; unsigned int u; } c; c.f = f;
    unsigned int u = c.u;
    u += 0x7FFFu + ((u >> 16) & 1u);       // round-to-nearest-even
    return (unsigned short)(u >> 16);
}

// Async global->LDS copy of 16B (+16B at offset:16). INST_OFFSET is added to
// both the LDS and the global address per the CDNA5 ISA (ch.10 async ops).
__device__ __forceinline__ void async_ld_lds_b128x2(unsigned lds_off, const void* gptr) {
    asm volatile("global_load_async_to_lds_b128 %0, %1, off"
                 :: "v"(lds_off), "v"(gptr) : "memory");
    asm volatile("global_load_async_to_lds_b128 %0, %1, off offset:16"
                 :: "v"(lds_off), "v"(gptr) : "memory");
}

__device__ __forceinline__ void wait_async0() {
#if __has_builtin(__builtin_amdgcn_s_wait_asynccnt)
    __builtin_amdgcn_s_wait_asynccnt(0);
#else
    asm volatile("s_wait_asynccnt 0x0" ::: "memory");
#endif
}

static constexpr int Bn = 8;     // batch
static constexpr int Nn = 4096;  // sequence
static constexpr int Dn = 512;   // feature

// ---------------------------------------------------------------------------
// f32 -> bf16 conversion (weights)
// ---------------------------------------------------------------------------
__global__ void cvt_bf16_kernel(const float* __restrict__ src,
                                unsigned short* __restrict__ dst, int n) {
    int i = blockIdx.x * blockDim.x + threadIdx.x;
    if (i < n) dst[i] = f2bf(src[i]);
}

// ---------------------------------------------------------------------------
// Projection: y[b, n, e] = x[b, n, :] . W[e, :] + bias[e]   (bf16 out)
// Workgroup: 512 threads = 16 waves, 64 rows x 512 cols of y.
// Wave (wy,wx): rows wy*16..+16, cols wx*128..+128 (8 accum frags)
// ---------------------------------------------------------------------------
__global__ __launch_bounds__(512) void proj_kernel(
    const float* __restrict__ x, const unsigned short* __restrict__ Wb,
    const float* __restrict__ bias, unsigned short* __restrict__ y) {
    __shared__ __align__(16) unsigned short xs[64][136]; // 64 rows x 128 k (bf16), padded

    const int tid = threadIdx.x;
    const int lane = tid & 31, wave = tid >> 5;
    const int wy = wave >> 2, wx = wave & 3;
    const int lane16 = lane & 15, laneHalf = lane >> 4;
    const int bb = blockIdx.y;
    const int rowBase = blockIdx.x * 64;

    const float* xb = x + (size_t)bb * Nn * Dn;
    unsigned short* yb = y + (size_t)bb * Nn * Dn;

    v8f acc[8];
#pragma unroll
    for (int c = 0; c < 8; ++c) acc[c] = (v8f){0,0,0,0,0,0,0,0};

    const int srow = tid >> 3;           // 0..63
    const int sseg = (tid & 7) * 16;     // 0..112

    for (int k0 = 0; k0 < Dn; k0 += 128) {
        // stage x[rowBase..+64, k0..+128] as bf16 into LDS
        {
            const float4* g = (const float4*)(xb + (size_t)(rowBase + srow) * Dn + k0 + sseg);
            alignas(16) unsigned short tmp[16];
#pragma unroll
            for (int i = 0; i < 4; ++i) {
                float4 f = g[i];
                tmp[4*i+0] = f2bf(f.x); tmp[4*i+1] = f2bf(f.y);
                tmp[4*i+2] = f2bf(f.z); tmp[4*i+3] = f2bf(f.w);
            }
            *(uint4*)&xs[srow][sseg]     = *(const uint4*)&tmp[0];
            *(uint4*)&xs[srow][sseg + 8] = *(const uint4*)&tmp[8];
        }
        __syncthreads();
#pragma unroll
        for (int k2 = 0; k2 < 128; k2 += 32) {
            BFrag A;
            const int arow = wy * 16 + lane16;
            const int ak = k2 + laneHalf * 16;
            *(uint4*)&A.h[0] = *(const uint4*)&xs[arow][ak];
            *(uint4*)&A.h[8] = *(const uint4*)&xs[arow][ak + 8];
#pragma unroll
            for (int c = 0; c < 8; ++c) {
                BFrag Bf;   // B[k][e] = W[e][k]: lane = col e, 16 contiguous k
                const int bcol = wx * 128 + c * 16 + lane16;
                const int bk = k0 + k2 + laneHalf * 16;
                const uint4* wp = (const uint4*)(Wb + (size_t)bcol * Dn + bk);
                *(uint4*)&Bf.h[0] = wp[0];
                *(uint4*)&Bf.h[8] = wp[1];
                acc[c] = __builtin_amdgcn_wmma_f32_16x16x32_bf16(
                    false, A.v, false, Bf.v, (short)0, acc[c], false, false);
            }
        }
        __syncthreads();
    }
    // bias + bf16 store
#pragma unroll
    for (int c = 0; c < 8; ++c) {
        const int col = wx * 128 + c * 16 + lane16;
        const float bv = bias[col];
        const int row0 = rowBase + wy * 16 + laneHalf * 8;
#pragma unroll
        for (int v = 0; v < 8; ++v)
            yb[(size_t)(row0 + v) * Dn + col] = f2bf(acc[c][v] + bv);
    }
}

// ---------------------------------------------------------------------------
// Fused: out[64 rows x 512] = sum_m (q1_tile @ q2_m^T) @ r_m
// 512 threads = 16 waves. Out acc: wave (wy,wx) owns rows wy*16, cols wx*128.
// Phase A: each wave computes one 16x16 tile of S (64x64), K-loop over 512,
//          with async global->LDS staging of the q1/q2 chunks.
// Phase B: S -> bf16 in LDS.
// Phase C: out_acc += S @ r, r staged transposed in LDS (aliases phase-A bufs).
// ---------------------------------------------------------------------------
__global__ __launch_bounds__(512) void fused_kernel(
    const unsigned short* __restrict__ q1, const unsigned short* __restrict__ q2,
    const unsigned short* __restrict__ r, float* __restrict__ out) {
    __shared__ __align__(16) union SMem {
        unsigned short ab[2][64][136];   // [0]=q1 chunk, [1]=q2 chunk (64 x 128 bf16)
        unsigned short rT[512][40];      // r transposed: [col][k], 512 x 32 bf16
    } smem;
    __shared__ __align__(16) unsigned short s_bf[64][72]; // S tile in bf16

    const int tid = threadIdx.x;
    const int lane = tid & 31, wave = tid >> 5;
    const int wy = wave >> 2, wx = wave & 3;
    const int lane16 = lane & 15, laneHalf = lane >> 4;
    const int bb = blockIdx.y;
    const int rowBase = blockIdx.x * 64;

    const unsigned short* q1b = q1 + (size_t)bb * Nn * Dn;
    const unsigned short* q2b = q2 + (size_t)bb * Nn * Dn;
    const unsigned short* rb  = r  + (size_t)bb * Nn * Dn;
    float* outb = out + (size_t)bb * Nn * Dn;

    v8f acc[8];
#pragma unroll
    for (int c = 0; c < 8; ++c) acc[c] = (v8f){0,0,0,0,0,0,0,0};

    const int srow = tid >> 3;           // staging row 0..63
    const int sseg = (tid & 7) * 16;     // staging k-segment

    for (int mb = 0; mb < Nn; mb += 64) {
        // prefetch next m-tile of q2/r into the cache hierarchy
        if (mb + 64 < Nn) {
            __builtin_prefetch(q2b + (size_t)(mb + 64 + srow) * Dn, 0, 1);
            __builtin_prefetch(rb  + (size_t)(mb + 64 + srow) * Dn, 0, 1);
        }
        // ---- Phase A: S(64x64) = q1[rowBase..] @ q2[mb..]^T --------------
        v8f s_acc = (v8f){0,0,0,0,0,0,0,0};
        for (int k0 = 0; k0 < Dn; k0 += 128) {
            {   // async global -> LDS staging (ASYNCcnt tracked)
                const unsigned lds_q1 = (unsigned)(unsigned long long)&smem.ab[0][srow][sseg];
                const unsigned lds_q2 = (unsigned)(unsigned long long)&smem.ab[1][srow][sseg];
                const unsigned short* g1 = q1b + (size_t)(rowBase + srow) * Dn + k0 + sseg;
                const unsigned short* g2 = q2b + (size_t)(mb + srow) * Dn + k0 + sseg;
                async_ld_lds_b128x2(lds_q1, g1);
                async_ld_lds_b128x2(lds_q2, g2);
                wait_async0();
            }
            __syncthreads();
#pragma unroll
            for (int k2 = 0; k2 < 128; k2 += 32) {
                BFrag A, Bf;
                const int arow = wy * 16 + lane16;
                const int ak = k2 + laneHalf * 16;
                *(uint4*)&A.h[0] = *(const uint4*)&smem.ab[0][arow][ak];
                *(uint4*)&A.h[8] = *(const uint4*)&smem.ab[0][arow][ak + 8];
                const int bcol = wx * 16 + lane16;          // m-row acting as B column
                const int bk = k2 + laneHalf * 16;
                *(uint4*)&Bf.h[0] = *(const uint4*)&smem.ab[1][bcol][bk];
                *(uint4*)&Bf.h[8] = *(const uint4*)&smem.ab[1][bcol][bk + 8];
                s_acc = __builtin_amdgcn_wmma_f32_16x16x32_bf16(
                    false, A.v, false, Bf.v, (short)0, s_acc, false, false);
            }
            __syncthreads();
        }
        // ---- Phase B: S -> bf16 in LDS -----------------------------------
        {
            const int sr = wy * 16 + laneHalf * 8;
            const int sc = wx * 16 + lane16;
#pragma unroll
            for (int v = 0; v < 8; ++v)
                s_bf[sr + v][sc] = f2bf(s_acc[v]);
        }
        __syncthreads();
        // ---- Phase C: out_acc += S @ r[mb..] -----------------------------
#pragma unroll
        for (int kk = 0; kk < 64; kk += 32) {
            // stage r rows [mb+kk, +32) transposed: rT[col][k].
            // 4 coalesced b128 row loads per thread, scattered b16 LDS stores.
#pragma unroll
            for (int i = 0; i < 4; ++i) {
                const int t2 = tid + i * 512;
                const int krow = t2 >> 6;          // 0..31
                const int cseg = (t2 & 63) * 8;    // 0..504
                uint4 d = *(const uint4*)(rb + (size_t)(mb + kk + krow) * Dn + cseg);
                alignas(16) unsigned short tmp[8];
                *(uint4*)&tmp[0] = d;
#pragma unroll
                for (int j = 0; j < 8; ++j)
                    smem.rT[cseg + j][krow] = tmp[j];
            }
            __syncthreads();
            BFrag A;
            const int arow = wy * 16 + lane16;
            const int ak = laneHalf * 16;
            *(uint4*)&A.h[0] = *(const uint4*)&s_bf[arow][kk + ak];
            *(uint4*)&A.h[8] = *(const uint4*)&s_bf[arow][kk + ak + 8];
#pragma unroll
            for (int c = 0; c < 8; ++c) {
                BFrag Bf;
                const int bcol = wx * 128 + c * 16 + lane16;
                const int bk = laneHalf * 16;
                *(uint4*)&Bf.h[0] = *(const uint4*)&smem.rT[bcol][bk];
                *(uint4*)&Bf.h[8] = *(const uint4*)&smem.rT[bcol][bk + 8];
                acc[c] = __builtin_amdgcn_wmma_f32_16x16x32_bf16(
                    false, A.v, false, Bf.v, (short)0, acc[c], false, false);
            }
            __syncthreads();
        }
    }
    // ---- epilogue: f32 store -------------------------------------------
#pragma unroll
    for (int c = 0; c < 8; ++c) {
        const int col = wx * 128 + c * 16 + lane16;
        const int row0 = rowBase + wy * 16 + laneHalf * 8;
#pragma unroll
        for (int v = 0; v < 8; ++v)
            outb[(size_t)(row0 + v) * Dn + col] = acc[c][v];
    }
}

// ---------------------------------------------------------------------------
// launch: inputs = x, Wq1_w, Wq1_b, Wq2_w, Wq2_b, WR_w, WR_b
// ws layout: q1 | q2 | r (bf16, B*N*D each) | Wq1b | Wq2b | WRb (bf16, D*D each)
//   total = (3*8*4096*512 + 3*512*512) * 2 bytes ~= 102 MB
// ---------------------------------------------------------------------------
extern "C" void kernel_launch(void* const* d_in, const int* in_sizes, int n_in,
                              void* d_out, int out_size, void* d_ws, size_t ws_size,
                              hipStream_t stream) {
    (void)in_sizes; (void)n_in; (void)out_size; (void)ws_size;

    const float* x   = (const float*)d_in[0];
    const float* W1  = (const float*)d_in[1];
    const float* b1  = (const float*)d_in[2];
    const float* W2  = (const float*)d_in[3];
    const float* b2  = (const float*)d_in[4];
    const float* WR  = (const float*)d_in[5];
    const float* bR  = (const float*)d_in[6];
    float* out = (float*)d_out;

    const size_t nd = (size_t)Bn * Nn * Dn;
    unsigned short* q1b = (unsigned short*)d_ws;
    unsigned short* q2b = q1b + nd;
    unsigned short* rb  = q2b + nd;
    unsigned short* w1b = rb + nd;
    unsigned short* w2b = w1b + (size_t)Dn * Dn;
    unsigned short* wrb = w2b + (size_t)Dn * Dn;

    const int wn = Dn * Dn;
    cvt_bf16_kernel<<<(wn + 255) / 256, 256, 0, stream>>>(W1, w1b, wn);
    cvt_bf16_kernel<<<(wn + 255) / 256, 256, 0, stream>>>(W2, w2b, wn);
    cvt_bf16_kernel<<<(wn + 255) / 256, 256, 0, stream>>>(WR, wrb, wn);

    dim3 grid(Nn / 64, Bn);
    proj_kernel<<<grid, 512, 0, stream>>>(x, w1b, b1, q1b);
    proj_kernel<<<grid, 512, 0, stream>>>(x, w2b, b2, q2b);
    proj_kernel<<<grid, 512, 0, stream>>>(x, wrb, bR, rb);

    fused_kernel<<<grid, 512, 0, stream>>>(q1b, q2b, rb, out);
}
